// Expert_layer2_1236950581358
// MI455X (gfx1250) — matrix operand order repaired and checked by reading, not compiled, not run on hardware
//
#include <hip/hip_runtime.h>
#include <hip/hip_bf16.h>

#define NB   4
#define KC   80
#define HW   4096
#define NTOK (NB*HW)
#define KP   96          // padded K dim (3 chunks of 32)
#define NKB  (HW/32)     // key blocks per batch (128)
#define EPSV 1e-5f

// gfx1250 async global->LDS path (ASYNCcnt), guarded so either toolchain compiles.
#if defined(__HIP_DEVICE_COMPILE__) && defined(__has_builtin)
#if __has_builtin(__builtin_amdgcn_global_load_async_to_lds_b128) && __has_builtin(__builtin_amdgcn_s_wait_asynccnt)
#define USE_ASYNC 1
#endif
#endif
#ifndef USE_ASYNC
#define USE_ASYNC 0
#endif

#if USE_ASYNC
typedef int v4ig __attribute__((vector_size(16)));   // matches builtin pointee type exactly
#define GPTR(p) ((__attribute__((address_space(1))) v4ig*)(p))
#define LPTR(p) ((__attribute__((address_space(3))) v4ig*)(p))
#endif

typedef __bf16         v16bf __attribute__((ext_vector_type(16)));
typedef __bf16         v8bf  __attribute__((ext_vector_type(8)));
typedef float          v8f   __attribute__((ext_vector_type(8)));
typedef unsigned short v8us  __attribute__((ext_vector_type(8)));

static __device__ __forceinline__ unsigned short f2bf(float f) {
  unsigned int u = __builtin_bit_cast(unsigned int, f);
  unsigned int r = u + 0x7fffu + ((u >> 16) & 1u);
  return (unsigned short)(r >> 16);
}

static __device__ __forceinline__ v8bf ldv8(const unsigned short* p) {
  return __builtin_bit_cast(v8bf, *(const v8us*)p);
}

static __device__ __forceinline__ v16bf cat16(v8bf a, v8bf b) {
  v16bf r;
#pragma unroll
  for (int i = 0; i < 8; ++i) { r[i] = a[i]; r[i + 8] = b[i]; }
  return r;
}

static __device__ __forceinline__ v8f zero8() {
  v8f r;
#pragma unroll
  for (int i = 0; i < 8; ++i) r[i] = 0.f;
  return r;
}

// ---------------------------------------------------------------- xmin
__global__ void k_xmin(const float* __restrict__ x2, float* __restrict__ out) {
  int i = blockIdx.x * blockDim.x + threadIdx.x;   // over NB*HW
  if (i >= NB * HW) return;
  int b = i / HW, hw = i % HW;
  const float* p = x2 + (size_t)b * KC * HW + hw;
  float m = p[0];
  for (int c = 1; c < KC; ++c) m = fminf(m, p[(size_t)c * HW]);
  out[i] = m;
}

// ---------------------------------------------------------------- direct 3x3 conv (pad=1)
__global__ void k_conv3x3(const float* __restrict__ in, const float* __restrict__ w,
                          float* __restrict__ out, int Cin, int Cout) {
  int i = blockIdx.x * 256 + threadIdx.x;
  int total = NB * Cout * HW;
  if (i >= total) return;
  int hw = i % HW; int t = i / HW; int o = t % Cout; int b = t / Cout;
  int y = hw >> 6, x = hw & 63;
  const float* ip = in + (size_t)b * Cin * HW;
  const float* wp = w + (size_t)o * Cin * 9;
  float acc = 0.f;
  for (int c = 0; c < Cin; ++c) {
    const float* icp = ip + (size_t)c * HW;
    const float* wcp = wp + c * 9;
    if (c + 1 < Cin) __builtin_prefetch(icp + HW + y * 64, 0, 1);
#pragma unroll
    for (int ky = 0; ky < 3; ++ky) {
      int yy = y + ky - 1; if ((unsigned)yy >= 64u) continue;
#pragma unroll
      for (int kx = 0; kx < 3; ++kx) {
        int xx = x + kx - 1; if ((unsigned)xx >= 64u) continue;
        acc += icp[yy * 64 + xx] * wcp[ky * 3 + kx];
      }
    }
  }
  out[i] = acc;
}

// ---------------------------------------------------------------- GroupNorm stats (block per (b,g))
__global__ void k_gn_stats(const float* __restrict__ x, float* __restrict__ mean,
                           float* __restrict__ rstd, int C, int Gp) {
  int bg = blockIdx.x; int g = bg % Gp; int b = bg / Gp;
  int cg = C / Gp;
  const float* p = x + ((size_t)b * C + (size_t)g * cg) * HW;
  int Ntot = cg * HW;
  float s = 0.f, s2 = 0.f;
  for (int i = threadIdx.x; i < Ntot; i += blockDim.x) { float v = p[i]; s += v; s2 += v * v; }
#pragma unroll
  for (int m = 16; m >= 1; m >>= 1) { s += __shfl_xor(s, m); s2 += __shfl_xor(s2, m); }
  __shared__ float sh0[8], sh1[8];
  int wid = threadIdx.x >> 5, lane = threadIdx.x & 31;
  if (lane == 0) { sh0[wid] = s; sh1[wid] = s2; }
  __syncthreads();
  if (threadIdx.x == 0) {
    float ts = 0.f, ts2 = 0.f;
    int nw = blockDim.x >> 5;
    for (int i = 0; i < nw; ++i) { ts += sh0[i]; ts2 += sh1[i]; }
    float m_ = ts / (float)Ntot;
    float v_ = ts2 / (float)Ntot - m_ * m_;
    mean[bg] = m_;
    rstd[bg] = rsqrtf(v_ + EPSV);
  }
}

// ---------------------------------------------------------------- GN apply (+ optional SiLU), in place
__global__ void k_gn_apply(float* __restrict__ x, const float* __restrict__ mean,
                           const float* __restrict__ rstd, const float* __restrict__ gamma,
                           const float* __restrict__ beta, int C, int Gp, int dosilu) {
  int i = blockIdx.x * 256 + threadIdx.x;
  int total = NB * C * HW;
  if (i >= total) return;
  int t = i / HW; int c = t % C; int b = t / C;
  int cg = C / Gp; int bg = b * Gp + c / cg;
  float v = (x[i] - mean[bg]) * rstd[bg] * gamma[c] + beta[c];
  if (dosilu) v = v / (1.f + __expf(-v));
  x[i] = v;
}

// ---------------------------------------------------------------- 1x1 conv (+bias, post: 0 none, 1 sigmoid)
__global__ void k_conv1x1(const float* __restrict__ in, const float* __restrict__ w,
                          const float* __restrict__ bias, float* __restrict__ out,
                          int Cin, int Cout, int post) {
  int i = blockIdx.x * 256 + threadIdx.x;
  int total = NB * Cout * HW;
  if (i >= total) return;
  int hw = i % HW; int t = i / HW; int o = t % Cout; int b = t / Cout;
  const float* ip = in + (size_t)b * Cin * HW + hw;
  const float* wp = w + (size_t)o * Cin;
  float acc = bias ? bias[o] : 0.f;
  for (int c = 0; c < Cin; ++c) acc += ip[(size_t)c * HW] * wp[c];
  if (post == 1) acc = 1.f / (1.f + __expf(-acc));
  out[i] = acc;
}

// ---------------------------------------------------------------- LayerNorm over channels -> bf16 [tok][96]
__global__ void k_ln_tok(const float* __restrict__ x, const float* __restrict__ g,
                         const float* __restrict__ be, unsigned short* __restrict__ out) {
  int wid = threadIdx.x >> 5, lane = threadIdx.x & 31;
  int t = blockIdx.x * 8 + wid;               // 2048 blocks * 8 waves = NTOK
  int b = t >> 12, hw = t & 4095;
  const float* p = x + (size_t)b * KC * HW + hw;
  float v0 = p[(size_t)lane * HW];
  float v1 = p[(size_t)(lane + 32) * HW];
  float v2 = (lane + 64 < KC) ? p[(size_t)(lane + 64) * HW] : 0.f;
  float s = v0 + v1 + v2, s2 = v0 * v0 + v1 * v1 + v2 * v2;
#pragma unroll
  for (int m = 16; m >= 1; m >>= 1) { s += __shfl_xor(s, m); s2 += __shfl_xor(s2, m); }
  float mean = s / (float)KC;
  float var = s2 / (float)KC - mean * mean;
  float r = rsqrtf(var + EPSV);
  unsigned short* op = out + (size_t)t * KP;
  op[lane]      = f2bf((v0 - mean) * r * g[lane] + be[lane]);
  op[lane + 32] = f2bf((v1 - mean) * r * g[lane + 32] + be[lane + 32]);
  op[lane + 64] = (lane + 64 < KC) ? f2bf((v2 - mean) * r * g[lane + 64] + be[lane + 64])
                                   : (unsigned short)0;
}

// ---------------------------------------------------------------- pack wq/wk/wv -> bf16 [96][96] each
__global__ void k_packw(const float* __restrict__ wq, const float* __restrict__ wk,
                        const float* __restrict__ wv, unsigned short* __restrict__ out) {
  int i = blockIdx.x * 256 + threadIdx.x;
  if (i >= 3 * KP * KP) return;
  int m = i / (KP * KP); int rem = i % (KP * KP); int o = rem / KP; int k = rem % KP;
  const float* w = (m == 0) ? wq : ((m == 1) ? wk : wv);
  float v = (o < KC && k < KC) ? w[o * KC + k] : 0.f;
  out[i] = f2bf(v);
}

// ---------------------------------------------------------------- WMMA GEMM: out = A[tok][96] x W^T
// vtmode=0: out[tok][96]. vtmode=1: block-transposed V layout [b][kb][d=96][key=32].
__global__ __launch_bounds__(128) void k_gemm(const unsigned short* __restrict__ A,
                                              const unsigned short* __restrict__ Wp,
                                              unsigned short* __restrict__ Out,
                                              const float* __restrict__ gate, int vtmode) {
  __shared__ __align__(16) unsigned short lA[64 * KP];
  __shared__ __align__(16) unsigned short lW[KP * KP];
  int tid = threadIdx.x;
  const uint4* ga = (const uint4*)(A + (size_t)blockIdx.x * 64 * KP);
  uint4* la4 = (uint4*)lA;
  for (int i = tid; i < 64 * KP / 8; i += 128) la4[i] = ga[i];
  const uint4* gw = (const uint4*)Wp;
  uint4* lw4 = (uint4*)lW;
  for (int i = tid; i < KP * KP / 8; i += 128) lw4[i] = gw[i];
  __syncthreads();

  int w = tid >> 5, lane = tid & 31, m = lane & 15, hi = lane >> 4;
  v8f acc[6];
#pragma unroll
  for (int i = 0; i < 6; ++i) acc[i] = zero8();

#pragma unroll
  for (int cc = 0; cc < 3; ++cc) {
    int c = cc * 32;
    int a0 = (w * 16 + m) * KP + c + hi * 8;
    v16bf av = cat16(ldv8(lA + a0), ldv8(lA + a0 + 16));
#pragma unroll
    for (int nt = 0; nt < 6; ++nt) {
      int b0 = (nt * 16 + m) * KP + c + hi * 8;
      v16bf bv = cat16(ldv8(lW + b0), ldv8(lW + b0 + 16));
      acc[nt] = __builtin_amdgcn_wmma_f32_16x16x32_bf16(false, av, false, bv,
                                                        (short)0, acc[nt], false, false);
    }
  }
#pragma unroll
  for (int nt = 0; nt < 6; ++nt) {
#pragma unroll
    for (int r = 0; r < 8; ++r) {
      int row = r + hi * 8;
      int col = nt * 16 + m;
      int tok = blockIdx.x * 64 + w * 16 + row;
      int b = tok >> 12, hw = tok & 4095;
      float v = acc[nt][r];
      if (gate && col < KC) {
        float gv = gate[((size_t)b * KC + col) * HW + hw];   // already sigmoid'ed
        v *= (1.f + 0.1f * gv);
      }
      if (vtmode) {
        int kb = hw >> 5, key = hw & 31;
        Out[(((size_t)b * NKB + kb) * KP + col) * 32 + key] = f2bf(v);
      } else {
        Out[(size_t)tok * KP + col] = f2bf(v);
      }
    }
  }
}

// ---------------------------------------------------------------- flash attention (bf16 WMMA, f32 accum)
// Double-buffered K/V tiles; async global->LDS when available, register pipeline otherwise.
__global__ __launch_bounds__(128) void k_attn(const unsigned short* __restrict__ Qb,
                                              const unsigned short* __restrict__ Kb,
                                              const unsigned short* __restrict__ Vt,
                                              float* __restrict__ Oout) {
  __shared__ __align__(16) unsigned short lQ[64 * KP];
  __shared__ __align__(16) unsigned short lK[2][32 * KP];
  __shared__ __align__(16) unsigned short lV[2][KP * 32];
  __shared__ __align__(16) unsigned short lP[4][16 * 32];
  const int TILE4 = 32 * KP / 8;   // 384 uint4 per tile (both K and V)
  int tid = threadIdx.x;
  int b = blockIdx.x >> 6;         // 64 query blocks per batch
  int qblk = blockIdx.x & 63;
  size_t qtok = ((size_t)b << 12) + (size_t)qblk * 64;

  const uint4* gq = (const uint4*)(Qb + qtok * KP);
  uint4* lq4 = (uint4*)lQ;
  for (int i = tid; i < 64 * KP / 8; i += 128) lq4[i] = gq[i];

  const uint4* kbase = (const uint4*)(Kb + ((size_t)b << 12) * KP);
  const uint4* vbase = (const uint4*)(Vt + ((size_t)b * NKB) * KP * 32);

  // prologue: tile 0 -> buffer 0
#if USE_ASYNC
  for (int i = tid; i < TILE4; i += 128) {
    __builtin_amdgcn_global_load_async_to_lds_b128(GPTR(kbase + i), LPTR((uint4*)lK[0] + i), 0, 0);
    __builtin_amdgcn_global_load_async_to_lds_b128(GPTR(vbase + i), LPTR((uint4*)lV[0] + i), 0, 0);
  }
  __builtin_amdgcn_s_wait_asynccnt(0);
#else
  for (int i = tid; i < TILE4; i += 128) {
    ((uint4*)lK[0])[i] = kbase[i];
    ((uint4*)lV[0])[i] = vbase[i];
  }
#endif
  __syncthreads();

  int w = tid >> 5, lane = tid & 31, m = lane & 15, hi = lane >> 4;
  v16bf qa[3];
#pragma unroll
  for (int c = 0; c < 3; ++c) {
    int a0 = (w * 16 + m) * KP + c * 32 + hi * 8;
    qa[c] = cat16(ldv8(lQ + a0), ldv8(lQ + a0 + 16));
  }

  v8f oacc[5];
#pragma unroll
  for (int i = 0; i < 5; ++i) oacc[i] = zero8();
  float mi[8], li[8];
#pragma unroll
  for (int r = 0; r < 8; ++r) { mi[r] = -1e30f; li[r] = 0.f; }
  const float scale = 0.1118033988749895f;   // 1/sqrt(80)

  for (int it = 0; it < NKB; ++it) {
    int cur = it & 1;
    // ---- issue prefetch of next tile into the other buffer
#if USE_ASYNC
    if (it + 1 < NKB) {
      int nxt = cur ^ 1;
      const uint4* ksrc = kbase + (size_t)(it + 1) * TILE4;
      const uint4* vsrc = vbase + (size_t)(it + 1) * TILE4;
      for (int i = tid; i < TILE4; i += 128) {
        __builtin_amdgcn_global_load_async_to_lds_b128(GPTR(ksrc + i), LPTR((uint4*)lK[nxt] + i), 0, 0);
        __builtin_amdgcn_global_load_async_to_lds_b128(GPTR(vsrc + i), LPTR((uint4*)lV[nxt] + i), 0, 0);
      }
    }
#else
    uint4 pk[3], pv[3];
    if (it + 1 < NKB) {
      const uint4* ksrc = kbase + (size_t)(it + 1) * TILE4;
      const uint4* vsrc = vbase + (size_t)(it + 1) * TILE4;
#pragma unroll
      for (int j = 0; j < 3; ++j) { pk[j] = ksrc[tid + j * 128]; pv[j] = vsrc[tid + j * 128]; }
    }
#endif
    const unsigned short* tK = lK[cur];
    const unsigned short* tV = lV[cur];

    // ---- S = Q K^T for 32 keys (two 16x16 tiles), 3 K-chunks each
    v8f s0 = zero8(), s1 = zero8();
#pragma unroll
    for (int c = 0; c < 3; ++c) {
      int b0 = m * KP + c * 32 + hi * 8;
      v16bf k0 = cat16(ldv8(tK + b0), ldv8(tK + b0 + 16));
      v16bf k1 = cat16(ldv8(tK + 16 * KP + b0), ldv8(tK + 16 * KP + b0 + 16));
      s0 = __builtin_amdgcn_wmma_f32_16x16x32_bf16(false, qa[c], false, k0, (short)0, s0, false, false);
      s1 = __builtin_amdgcn_wmma_f32_16x16x32_bf16(false, qa[c], false, k1, (short)0, s1, false, false);
    }

    // ---- online softmax over this 32-key block (row = r + 8*hi, col = lane%16)
    unsigned short* lp = lP[w];
#pragma unroll
    for (int r = 0; r < 8; ++r) {
      float a = s0[r] * scale, bb = s1[r] * scale;
      float mx = fmaxf(a, bb);
#pragma unroll
      for (int xm = 8; xm >= 1; xm >>= 1) mx = fmaxf(mx, __shfl_xor(mx, xm));
      float mnew = fmaxf(mi[r], mx);
      float alpha = __expf(mi[r] - mnew);
      float p0 = __expf(a - mnew), p1 = __expf(bb - mnew);
      float ps = p0 + p1;
#pragma unroll
      for (int xm = 8; xm >= 1; xm >>= 1) ps += __shfl_xor(ps, xm);
      li[r] = li[r] * alpha + ps;
      mi[r] = mnew;
#pragma unroll
      for (int dt = 0; dt < 5; ++dt) oacc[dt][r] *= alpha;
      int rowm = r + hi * 8;
      lp[rowm * 32 + m]      = f2bf(p0);
      lp[rowm * 32 + 16 + m] = f2bf(p1);
    }

    // ---- O += P V : A = P (16x32), B = V^T rows (contiguous in tV)
    int pa0 = m * 32 + hi * 8;
    v16bf pav = cat16(ldv8(lp + pa0), ldv8(lp + pa0 + 16));
#pragma unroll
    for (int dt = 0; dt < 5; ++dt) {
      int b0 = (dt * 16 + m) * 32 + hi * 8;
      v16bf vv = cat16(ldv8(tV + b0), ldv8(tV + b0 + 16));
      oacc[dt] = __builtin_amdgcn_wmma_f32_16x16x32_bf16(false, pav, false, vv, (short)0, oacc[dt], false, false);
    }

    // ---- publish next tile
#if USE_ASYNC
    __builtin_amdgcn_s_wait_asynccnt(0);
#else
    if (it + 1 < NKB) {
      int nxt = cur ^ 1;
#pragma unroll
      for (int j = 0; j < 3; ++j) {
        ((uint4*)lK[nxt])[tid + j * 128] = pk[j];
        ((uint4*)lV[nxt])[tid + j * 128] = pv[j];
      }
    }
#endif
    __syncthreads();
  }

#pragma unroll
  for (int r = 0; r < 8; ++r) {
    float inv = 1.f / li[r];
    int rowm = r + hi * 8;
    size_t tok = qtok + (size_t)(w * 16 + rowm);
#pragma unroll
    for (int dt = 0; dt < 5; ++dt)
      Oout[tok * KC + dt * 16 + m] = oacc[dt][r] * inv;
  }
}

// ---------------------------------------------------------------- (attn + resid) -> LN -> channel-major map
__global__ void k_add_ln(const float* __restrict__ attn, const float* __restrict__ resid,
                         const float* __restrict__ g, const float* __restrict__ be,
                         float* __restrict__ ymap) {
  int wid = threadIdx.x >> 5, lane = threadIdx.x & 31;
  int t = blockIdx.x * 8 + wid;
  int b = t >> 12, hw = t & 4095;
  const float* ap = attn + (size_t)t * KC;
  const float* rp = resid + (size_t)b * KC * HW + hw;
  float v0 = ap[lane]      + rp[(size_t)lane * HW];
  float v1 = ap[lane + 32] + rp[(size_t)(lane + 32) * HW];
  float v2 = (lane + 64 < KC) ? ap[lane + 64] + rp[(size_t)(lane + 64) * HW] : 0.f;
  float s = v0 + v1 + v2, s2 = v0 * v0 + v1 * v1 + v2 * v2;
#pragma unroll
  for (int m = 16; m >= 1; m >>= 1) { s += __shfl_xor(s, m); s2 += __shfl_xor(s2, m); }
  float mean = s / (float)KC;
  float var = s2 / (float)KC - mean * mean;
  float r = rsqrtf(var + EPSV);
  float* yp = ymap + (size_t)b * KC * HW + hw;
  yp[(size_t)lane * HW]        = (v0 - mean) * r * g[lane] + be[lane];
  yp[(size_t)(lane + 32) * HW] = (v1 - mean) * r * g[lane + 32] + be[lane + 32];
  if (lane + 64 < KC)
    yp[(size_t)(lane + 64) * HW] = (v2 - mean) * r * g[lane + 64] + be[lane + 64];
}

// ================================================================ host
extern "C" void kernel_launch(void* const* d_in, const int* in_sizes, int n_in,
                              void* d_out, int out_size, void* d_ws, size_t ws_size,
                              hipStream_t stream) {
  (void)in_sizes; (void)n_in; (void)out_size; (void)ws_size;
  const float* x1    = (const float*)d_in[0];
  const float* x2    = (const float*)d_in[1];
  const float* w_p1  = (const float*)d_in[2];
  const float* gn1_g = (const float*)d_in[3];
  const float* gn1_b = (const float*)d_in[4];
  const float* w_id  = (const float*)d_in[5];
  const float* gnid_g= (const float*)d_in[6];
  const float* gnid_b= (const float*)d_in[7];
  const float* wq    = (const float*)d_in[8];
  const float* wk    = (const float*)d_in[9];
  const float* wv    = (const float*)d_in[10];
  const float* wg1   = (const float*)d_in[11];
  const float* gng_g = (const float*)d_in[12];
  const float* gng_b = (const float*)d_in[13];
  const float* wg2   = (const float*)d_in[14];
  const float* bg2   = (const float*)d_in[15];
  const float* lnx1_g= (const float*)d_in[16];
  const float* lnx1_b= (const float*)d_in[17];
  const float* lnx2_g= (const float*)d_in[18];
  const float* lnx2_b= (const float*)d_in[19];
  const float* lno_g = (const float*)d_in[20];
  const float* lno_b = (const float*)d_in[21];
  const float* wh1   = (const float*)d_in[22];
  const float* gnh_g = (const float*)d_in[23];
  const float* gnh_b = (const float*)d_in[24];
  const float* wh2   = (const float*)d_in[25];
  const float* bh2   = (const float*)d_in[26];
  float* out = (float*)d_out;

  char* ws = (char*)d_ws;
  size_t off = 0;
  auto carve = [&](size_t bytes) -> char* {
    char* p = ws + off;
    off = (off + bytes + 255) & ~(size_t)255;
    return p;
  };
  float* bufA  = (float*)carve((size_t)NB * KC * HW * 4);   // x1 conv / x1p ; later head mid
  float* bufB  = (float*)carve((size_t)NB * KC * HW * 4);   // x2_id (GN)
  float* bufC  = (float*)carve((size_t)NB * KC * HW * 4);   // gate g ; later y_map
  float* buf40 = (float*)carve((size_t)NB * 40 * HW * 4);   // gate mid
  float* stm   = (float*)carve(256);
  float* str   = (float*)carve(256);
  unsigned short* x1s = (unsigned short*)carve((size_t)NTOK * KP * 2);
  unsigned short* x2s = (unsigned short*)carve((size_t)NTOK * KP * 2);
  unsigned short* qb  = (unsigned short*)carve((size_t)NTOK * KP * 2);
  unsigned short* kb  = (unsigned short*)carve((size_t)NTOK * KP * 2);
  unsigned short* vt  = (unsigned short*)carve((size_t)NTOK * KP * 2); // block-transposed V
  unsigned short* wpk = (unsigned short*)carve((size_t)3 * KP * KP * 2);
  float* attn_o = (float*)carve((size_t)NTOK * KC * 4);

  // xmin output (second half of d_out)
  k_xmin<<<(NB * HW + 255) / 256, 256, 0, stream>>>(x2, out + NB * HW);

  // proj_x1: conv3x3 + GN + SiLU
  k_conv3x3<<<NB * KC * HW / 256, 256, 0, stream>>>(x1, w_p1, bufA, 64, KC);
  k_gn_stats<<<NB * 5, 256, 0, stream>>>(bufA, stm, str, KC, 5);
  k_gn_apply<<<NB * KC * HW / 256, 256, 0, stream>>>(bufA, stm, str, gn1_g, gn1_b, KC, 5, 1);

  // sequences + LN -> bf16
  k_ln_tok<<<NTOK / 8, 256, 0, stream>>>(bufA, lnx1_g, lnx1_b, x1s);
  k_ln_tok<<<NTOK / 8, 256, 0, stream>>>(x2,   lnx2_g, lnx2_b, x2s);

  // x2_id: conv1x1 + GN
  k_conv1x1<<<NB * KC * HW / 256, 256, 0, stream>>>(x2, w_id, nullptr, bufB, KC, KC, 0);
  k_gn_stats<<<NB * 5, 256, 0, stream>>>(bufB, stm, str, KC, 5);
  k_gn_apply<<<NB * KC * HW / 256, 256, 0, stream>>>(bufB, stm, str, gnid_g, gnid_b, KC, 5, 0);

  // gate branch: conv3x3 -> GN -> SiLU -> conv1x1 + bias -> sigmoid
  k_conv3x3<<<NB * 40 * HW / 256, 256, 0, stream>>>(x2, wg1, buf40, KC, 40);
  k_gn_stats<<<NB * 5, 256, 0, stream>>>(buf40, stm, str, 40, 5);
  k_gn_apply<<<NB * 40 * HW / 256, 256, 0, stream>>>(buf40, stm, str, gng_g, gng_b, 40, 5, 1);
  k_conv1x1<<<NB * KC * HW / 256, 256, 0, stream>>>(buf40, wg2, bg2, bufC, 40, KC, 1);

  // q,k,v WMMA GEMMs (v gated + block-transposed in epilogue)
  k_packw<<<(3 * KP * KP + 255) / 256, 256, 0, stream>>>(wq, wk, wv, wpk);
  k_gemm<<<NTOK / 64, 128, 0, stream>>>(x1s, wpk,               qb, nullptr, 0);
  k_gemm<<<NTOK / 64, 128, 0, stream>>>(x2s, wpk + KP * KP,     kb, nullptr, 0);
  k_gemm<<<NTOK / 64, 128, 0, stream>>>(x2s, wpk + 2 * KP * KP, vt, bufC,    1);

  // flash attention (double-buffered, pipelined)
  k_attn<<<NB * (HW / 64), 128, 0, stream>>>(qb, kb, vt, attn_o);

  // residual + LN -> channel-major y_map (reuse bufC)
  k_add_ln<<<NTOK / 8, 256, 0, stream>>>(attn_o, bufB, lno_g, lno_b, bufC);

  // head: conv3x3 -> GN(GH=5) -> SiLU -> conv1x1 + bias -> sigmoid -> pred
  k_conv3x3<<<NB * KC * HW / 256, 256, 0, stream>>>(bufC, wh1, bufA, KC, KC);
  k_gn_stats<<<NB * 5, 256, 0, stream>>>(bufA, stm, str, KC, 5);
  k_gn_apply<<<NB * KC * HW / 256, 256, 0, stream>>>(bufA, stm, str, gnh_g, gnh_b, KC, 5, 1);
  k_conv1x1<<<NB * 1 * HW / 256, 256, 0, stream>>>(bufA, wh2, bh2, out, KC, 1, 1);
}
